// SlicedExpertManagerFusedKernel_86294482911902
// MI455X (gfx1250) — compile-verified, matrix operand order checked by reading, not run on hardware
//
#include <hip/hip_runtime.h>
#include <hip/hip_bf16.h>

// MoE expert FFN: per expert e, out_e = relu(x_e @ wi_e) @ wo_e
// E=8, T=4096, d_model=1024, d_ff=4096. Compute-bound (550 GFLOP vs ~0.5GB HBM),
// so run the GEMMs in bf16 via v_wmma_f32_16x16x32_bf16 (f32 accumulate).
//
// Pass 0: transpose+convert wi -> wiT [E, d_ff, d_model] bf16,
//                           wo -> woT [E, d_model, d_ff] bf16   (in d_ws)
// Pass 1: H = relu(x @ wi) -> bf16 in d_ws (B read from wiT, copy-only staging)
// Pass 2: out = H @ wo     -> f32 in d_out (A and B both bf16, copy-only staging)
// Workspace requirement: 256MB (H) + 64MB (wiT) + 64MB (woT) = 384 MB.
// f32->bf16 uses vector fptrunc so the backend emits packed v_cvt_pk_bf16_f32.

typedef __attribute__((ext_vector_type(16))) __bf16        v16bf;
typedef __attribute__((ext_vector_type(8)))  float         v8f;
typedef __attribute__((ext_vector_type(4)))  unsigned int  u32x4;
typedef __attribute__((ext_vector_type(2)))  float         v2f;
typedef __attribute__((ext_vector_type(2)))  __bf16        v2bf;

#define BM 128
#define BN 128
#define BK 32
#define LDS_STRIDE 40   // ushorts per row = 80 bytes: 16B-aligned + bank-conflict-free
#define THREADS 256

// single f32 -> bf16 (epilogue / transpose; 1 cvt op)
__device__ __forceinline__ unsigned short bf1(float a) {
  union { __bf16 h; unsigned short u; } r;
  r.h = (__bf16)a;
  return r.u;
}
// packed pair f32 -> bf16x2: vector fptrunc -> single v_cvt_pk_bf16_f32
__device__ __forceinline__ unsigned int pkbf2(float a, float b) {
  v2f v;
  v[0] = a;
  v[1] = b;
  v2bf h = __builtin_convertvector(v, v2bf);
  union { v2bf h; unsigned int u; } r;
  r.h = h;
  return r.u;
}

union Frag { u32x4 q[2]; v16bf v; };

// ---------------- Pass 0: tiled transpose + f32->bf16 convert ----------------
// in:  [E, R, C] f32 row-major   out: [E, C, R] bf16 row-major
__global__ __launch_bounds__(256)
void transpose_cvt(const float* __restrict__ in, unsigned short* __restrict__ out,
                   int R, int C)
{
  __shared__ float tile[32][33];
  const int t  = threadIdx.x;
  const int tx = t & 31, ty = t >> 5;
  const size_t batch = (size_t)R * C;
  const float*    src = in  + (size_t)blockIdx.z * batch;
  unsigned short* dst = out + (size_t)blockIdx.z * batch;
  const int bx = blockIdx.x * 32;  // col base in input
  const int by = blockIdx.y * 32;  // row base in input
  #pragma unroll
  for (int i = 0; i < 4; ++i)
    tile[ty + i * 8][tx] = src[(size_t)(by + ty + i * 8) * C + bx + tx];
  __syncthreads();
  #pragma unroll
  for (int i = 0; i < 4; ++i)
    dst[(size_t)(bx + ty + i * 8) * R + by + tx] = bf1(tile[tx][ty + i * 8]);
}

// ---------------- Main GEMM: C = op(A) @ B, B pre-transposed bf16 ----------------
template<bool A_IS_BF16, bool RELU_TO_BF16>
__global__ __launch_bounds__(THREADS)
void moe_gemm(const void* __restrict__ Abase,            // [E, M, K] f32 or bf16
              const unsigned short* __restrict__ BTbase, // [E, N, K] bf16 (pre-transposed)
              void* __restrict__ Obase,                  // [E, M, N] bf16(relu) or f32
              int K, int N)
{
  __shared__ __align__(16) unsigned short As[2][BM * LDS_STRIDE];
  __shared__ __align__(16) unsigned short Bs[2][BN * LDS_STRIDE];

  const int t    = threadIdx.x;
  const int lane = t & 31;
  const int wave = t >> 5;
  const int e    = blockIdx.z;
  const int m0   = blockIdx.y * BM;
  const int n0   = blockIdx.x * BN;
  const int M    = gridDim.y * BM;

  const size_t aBatch = (size_t)M * K;
  const size_t bBatch = (size_t)K * N;
  const size_t oBatch = (size_t)M * N;

  const float*          Af = ((const float*)Abase)          + (size_t)e * aBatch;
  const unsigned short* Ah = ((const unsigned short*)Abase) + (size_t)e * aBatch;
  const unsigned short* Bt = BTbase                         + (size_t)e * bBatch;

  // wave -> 32x64 sub-tile of the 128x128 macro tile
  const int wm    = (wave & 3) * 32;   // 4 M groups
  const int wn    = (wave >> 2) * 64;  // 2 N groups
  const int lhalf = lane >> 4;         // 0/1: which K chunk this lane holds
  const int lrow  = lane & 15;         // row (A) / col (B) within 16-tile

  v8f acc[2][4];
  #pragma unroll
  for (int i = 0; i < 2; ++i)
    #pragma unroll
    for (int j = 0; j < 4; ++j)
      #pragma unroll
      for (int r = 0; r < 8; ++r)
        acc[i][j][r] = 0.0f;

  // ---- staging helpers (global -> regs, regs -> LDS) ----
  float4 paf[4];  // A prefetch, f32 path
  u32x4  pab[2];  // A prefetch, bf16 path
  u32x4  pb[2];   // B prefetch (always bf16)

  auto loadA = [&](int k0) {
    if (A_IS_BF16) {
      #pragma unroll
      for (int it = 0; it < 2; ++it) {
        int idx = t + it * THREADS;  int r = idx >> 2;  int c = (idx & 3) << 3;
        pab[it] = *(const u32x4*)(Ah + (size_t)(m0 + r) * K + k0 + c);
      }
    } else {
      #pragma unroll
      for (int it = 0; it < 4; ++it) {
        int idx = t + it * THREADS;  int r = idx >> 3;  int c = (idx & 7) << 2;
        paf[it] = *(const float4*)(Af + (size_t)(m0 + r) * K + k0 + c);
      }
    }
  };
  auto loadB = [&](int k0) {
    #pragma unroll
    for (int it = 0; it < 2; ++it) {
      int idx = t + it * THREADS;  int r = idx >> 2;  int c = (idx & 3) << 3;
      pb[it] = *(const u32x4*)(Bt + (size_t)(n0 + r) * K + k0 + c);
    }
  };
  auto storeA = [&](unsigned short* dst) {
    if (A_IS_BF16) {
      #pragma unroll
      for (int it = 0; it < 2; ++it) {
        int idx = t + it * THREADS;  int r = idx >> 2;  int c = (idx & 3) << 3;
        *(u32x4*)(dst + r * LDS_STRIDE + c) = pab[it];
      }
    } else {
      #pragma unroll
      for (int it = 0; it < 4; ++it) {
        int idx = t + it * THREADS;  int r = idx >> 3;  int c = (idx & 7) << 2;
        const float4 v = paf[it];
        unsigned long long pk =
             (unsigned long long)pkbf2(v.x, v.y)
          | ((unsigned long long)pkbf2(v.z, v.w) << 32);
        *(unsigned long long*)(dst + r * LDS_STRIDE + c) = pk;
      }
    }
  };
  auto storeB = [&](unsigned short* dst) {
    #pragma unroll
    for (int it = 0; it < 2; ++it) {
      int idx = t + it * THREADS;  int r = idx >> 2;  int c = (idx & 3) << 3;
      *(u32x4*)(dst + r * LDS_STRIDE + c) = pb[it];
    }
  };

  // ---- pipelined main loop: double-buffered LDS ----
  const int NK = K / BK;
  loadA(0); loadB(0);
  storeA(As[0]); storeB(Bs[0]);
  __syncthreads();

  for (int kt = 0; kt < NK; ++kt) {
    const int cur = kt & 1;
    const bool more = (kt + 1 < NK);

    if (more) { loadA((kt + 1) * BK); loadB((kt + 1) * BK); }  // overlap with WMMA

    // fragment loads: two 16B LDS reads per fragment, per ISA 16-bit layout
    // (lanes 0-15 hold K 0..7 & 16..23; lanes 16-31 hold K 8..15 & 24..31)
    Frag a[2], b[4];
    #pragma unroll
    for (int i = 0; i < 2; ++i) {
      const unsigned short* p = &As[cur][(wm + i * 16 + lrow) * LDS_STRIDE + (lhalf << 3)];
      a[i].q[0] = *(const u32x4*)(p);
      a[i].q[1] = *(const u32x4*)(p + 16);
    }
    #pragma unroll
    for (int j = 0; j < 4; ++j) {
      const unsigned short* p = &Bs[cur][(wn + j * 16 + lrow) * LDS_STRIDE + (lhalf << 3)];
      b[j].q[0] = *(const u32x4*)(p);
      b[j].q[1] = *(const u32x4*)(p + 16);
    }

    #pragma unroll
    for (int i = 0; i < 2; ++i)
      #pragma unroll
      for (int j = 0; j < 4; ++j)
        acc[i][j] = __builtin_amdgcn_wmma_f32_16x16x32_bf16(
            false, a[i].v, false, b[j].v, (short)0, acc[i][j], false, false);

    if (more) { storeA(As[cur ^ 1]); storeB(Bs[cur ^ 1]); }
    __syncthreads();
  }

  // ---- epilogue ----
  // C/D f32 16x16 layout: VGPR r holds (M = r + 8*lhalf, N = lrow)
  if (RELU_TO_BF16) {
    unsigned short* O = ((unsigned short*)Obase) + (size_t)e * oBatch;
    #pragma unroll
    for (int i = 0; i < 2; ++i)
      #pragma unroll
      for (int j = 0; j < 4; ++j)
        #pragma unroll
        for (int r = 0; r < 8; ++r) {
          int gm = m0 + wm + i * 16 + lhalf * 8 + r;
          int gn = n0 + wn + j * 16 + lrow;
          O[(size_t)gm * N + gn] = bf1(fmaxf(acc[i][j][r], 0.0f));
        }
  } else {
    float* O = ((float*)Obase) + (size_t)e * oBatch;
    #pragma unroll
    for (int i = 0; i < 2; ++i)
      #pragma unroll
      for (int j = 0; j < 4; ++j)
        #pragma unroll
        for (int r = 0; r < 8; ++r) {
          int gm = m0 + wm + i * 16 + lhalf * 8 + r;
          int gn = n0 + wn + j * 16 + lrow;
          O[(size_t)gm * N + gn] = acc[i][j][r];
        }
  }
}

extern "C" void kernel_launch(void* const* d_in, const int* in_sizes, int n_in,
                              void* d_out, int out_size, void* d_ws, size_t ws_size,
                              hipStream_t stream) {
  (void)in_sizes; (void)n_in; (void)out_size; (void)ws_size;

  const int E = 8, T = 4096, DM = 1024, DF = 4096;

  const float* x  = (const float*)d_in[0];   // [E*T, DM]
  const float* wi = (const float*)d_in[1];   // [E, DM, DF]
  const float* wo = (const float*)d_in[2];   // [E, DF, DM]
  float* out = (float*)d_out;                // [E*T, DM]

  // workspace layout (bf16 elements): H | wiT | woT   -> 384 MB total
  unsigned short* H   = (unsigned short*)d_ws;                       // [E, T, DF]
  unsigned short* wiT = H   + (size_t)E * T * DF;                    // [E, DF, DM]
  unsigned short* woT = wiT + (size_t)E * DF * DM;                   // [E, DM, DF]

  dim3 blk(THREADS);

  // Pass 0: weight transpose + convert
  transpose_cvt<<<dim3(DF / 32, DM / 32, E), blk, 0, stream>>>(wi, wiT, DM, DF);
  transpose_cvt<<<dim3(DM / 32, DF / 32, E), blk, 0, stream>>>(wo, woT, DF, DM);

  // Pass 1: H[e] = relu(x[e] @ wi[e])   M=T, K=DM, N=DF
  dim3 g1(DF / BN, T / BM, E);
  moe_gemm<false, true><<<g1, blk, 0, stream>>>(x, wiT, (void*)H, DM, DF);

  // Pass 2: out[e] = H[e] @ wo[e]       M=T, K=DF, N=DM
  dim3 g2(DM / BN, T / BM, E);
  moe_gemm<true, false><<<g2, blk, 0, stream>>>((const void*)H, woT, (void*)out, DF, DM);
}
